// LSTM_558345748830
// MI455X (gfx1250) — matrix operand order repaired and checked by reading, not compiled, not run on hardware
//
#include <hip/hip_runtime.h>

// ---- CDNA5 (gfx1250) fused 2-layer LSTM scan using v_wmma_f32_16x16x32_f16 ----
// One wave32 owns 16 batch rows; both layers' h/c state + all recurrent weight
// fragments live in VGPRs; h round-trips C-layout -> LDS -> A-layout per step.

typedef __attribute__((ext_vector_type(16))) _Float16 v16h;
typedef __attribute__((ext_vector_type(8)))  float    v8f;
typedef __attribute__((ext_vector_type(4)))  float    v4f;

union Frag16 { v16h v; _Float16 h[16]; v4f q[2]; };

#define T_STEPS 512
#define WAVES   4

__device__ __forceinline__ float fast_sigmoid(float x) {
    return __builtin_amdgcn_rcpf(1.0f + __expf(-x));
}
__device__ __forceinline__ float fast_tanh(float x) {
    float e = __expf(2.0f * x);                 // inf-safe: x>>0 -> 1, x<<0 -> -1
    return 1.0f - 2.0f * __builtin_amdgcn_rcpf(e + 1.0f);
}

__global__ __launch_bounds__(WAVES * 32)
void lstm_fused_wmma(const float* __restrict__ x,
                     const float* __restrict__ w_ih0, const float* __restrict__ w_hh0,
                     const float* __restrict__ b_ih0, const float* __restrict__ b_hh0,
                     const float* __restrict__ w_ih1, const float* __restrict__ w_hh1,
                     const float* __restrict__ b_ih1, const float* __restrict__ b_hh1,
                     const float* __restrict__ fc_w,  const float* __restrict__ fc_b,
                     float* __restrict__ out)
{
    __shared__ _Float16 hbuf[WAVES][16 * 32];   // 16 rows x 32 hidden, f16, row-major
    __shared__ float    xbuf[WAVES][16];
    __shared__ float    oacc[WAVES][16];

    const int lane    = threadIdx.x & 31;
    const int w       = threadIdx.x >> 5;
    const int tile    = blockIdx.x * WAVES + w;
    const int rowbase = tile * 16;

    const int nlo   = lane & 15;   // column-in-tile (N) / row-in-tile (M) selector
    const int khalf = lane >> 4;   // lane-half selects K sub-range in A/B layouts

    // ---- per-lane gate-column constants (C/D layout: N = 16*tile + nlo) ----
    float w0c[8], b0c[8], b1c[8];
    #pragma unroll
    for (int n = 0; n < 8; ++n) {
        int g = 16 * n + nlo;
        w0c[n] = w_ih0[g];                       // w_ih_0 is [128,1]
        b0c[n] = b_ih0[g] + b_hh0[g];
        b1c[n] = b_ih1[g] + b_hh1[g];
    }
    const float fcw0 = fc_w[nlo];
    const float fcw1 = fc_w[16 + nlo];

    // ---- preload B fragments (f32 weights -> f16 WMMA B layout) ----
    // B tile n covers gate cols [16n,16n+16); lane: N=nlo, slot s<8 -> K=8*khalf+s, s>=8 -> K=16+8*khalf+(s-8).
    // B[k][col] = w[col][k]  (gates = h @ w^T), w row-major [128,32].
    Frag16 Bhh0[8], Bih1[8], Bhh1[8];
    #pragma unroll
    for (int n = 0; n < 8; ++n) {
        const int off = (16 * n + nlo) * 32 + 8 * khalf;
        const float* p0 = w_hh0 + off;
        const float* p1 = w_ih1 + off;
        const float* p2 = w_hh1 + off;
        #pragma unroll
        for (int s = 0; s < 16; ++s) {
            Bhh0[n].h[s] = (_Float16)p0[s + (s & 8)];
            Bih1[n].h[s] = (_Float16)p1[s + (s & 8)];
            Bhh1[n].h[s] = (_Float16)p2[s + (s & 8)];
        }
    }

    // ---- recurrent state (registers only) ----
    v16h h0A = {};
    v16h h1A = {};
    float c0v[16], c1v[16], hl1[16];
    #pragma unroll
    for (int i = 0; i < 16; ++i) { c0v[i] = 0.f; c1v[i] = 0.f; hl1[i] = 0.f; }

    const float* xrow = x + (size_t)(rowbase + nlo) * T_STEPS;

    #pragma unroll 1
    for (int t = 0; t < T_STEPS; ++t) {
        // ---- stage x[:, t] for this 16-row tile (I == 1: rank-1 projection) ----
        if (lane < 16) {
            xbuf[w][lane] = xrow[t];
            __builtin_prefetch(xrow + (t + 24 < T_STEPS ? t + 24 : T_STEPS - 1), 0, 1);
        }
        asm volatile("s_wait_dscnt 0" ::: "memory");
        v4f xq0 = *(const v4f*)&xbuf[w][8 * khalf];
        v4f xq1 = *(const v4f*)&xbuf[w][8 * khalf + 4];
        float xr[8] = { xq0[0], xq0[1], xq0[2], xq0[3], xq1[0], xq1[1], xq1[2], xq1[3] };

        // ---- layer 0: gates = bias + x*w_ih0 (C init) + h0 @ w_hh0^T (WMMA) ----
        v8f acc[8];
        #pragma unroll
        for (int n = 0; n < 8; ++n) {
            #pragma unroll
            for (int v2 = 0; v2 < 8; ++v2)
                acc[n][v2] = b0c[n] + xr[v2] * w0c[n];
        }
        #pragma unroll
        for (int n = 0; n < 8; ++n)
            acc[n] = __builtin_amdgcn_wmma_f32_16x16x32_f16(
                         false, h0A, false, Bhh0[n].v, (short)0, acc[n], false, false);

        // ---- layer 0 elementwise; h0_t -> LDS as f16 (row-major 16x32) ----
        #pragma unroll
        for (int nh = 0; nh < 2; ++nh) {
            #pragma unroll
            for (int v2 = 0; v2 < 8; ++v2) {
                float iv = fast_sigmoid(acc[0 + nh][v2]);
                float fv = fast_sigmoid(acc[2 + nh][v2]);
                float gv = fast_tanh  (acc[4 + nh][v2]);
                float ov = fast_sigmoid(acc[6 + nh][v2]);
                float c  = fv * c0v[nh * 8 + v2] + iv * gv;
                c0v[nh * 8 + v2] = c;
                float hv = ov * fast_tanh(c);
                int M = v2 + 8 * khalf;                      // C layout: M = vgpr + 8*(lane>=16)
                hbuf[w][M * 32 + 16 * nh + nlo] = (_Float16)hv;
            }
        }
        asm volatile("s_wait_dscnt 0" ::: "memory");

        // A-layout reload: lane M=nlo; slots 0-7 -> K=8*khalf.., slots 8-15 -> K=16+8*khalf..
        Frag16 a0;
        {
            const _Float16* hp = &hbuf[w][nlo * 32];
            a0.q[0] = *(const v4f*)(hp + 8 * khalf);
            a0.q[1] = *(const v4f*)(hp + 16 + 8 * khalf);
        }

        // ---- layer 1: gates = bias + h0_t @ w_ih1^T + h1 @ w_hh1^T ----
        #pragma unroll
        for (int n = 0; n < 8; ++n) {
            v8f a;
            #pragma unroll
            for (int v2 = 0; v2 < 8; ++v2) a[v2] = b1c[n];
            a = __builtin_amdgcn_wmma_f32_16x16x32_f16(
                    false, a0.v, false, Bih1[n].v, (short)0, a, false, false);
            acc[n] = __builtin_amdgcn_wmma_f32_16x16x32_f16(
                    false, h1A, false, Bhh1[n].v, (short)0, a, false, false);
        }

        // ---- layer 1 elementwise; h1_t -> LDS as f16 ----
        #pragma unroll
        for (int nh = 0; nh < 2; ++nh) {
            #pragma unroll
            for (int v2 = 0; v2 < 8; ++v2) {
                float iv = fast_sigmoid(acc[0 + nh][v2]);
                float fv = fast_sigmoid(acc[2 + nh][v2]);
                float gv = fast_tanh  (acc[4 + nh][v2]);
                float ov = fast_sigmoid(acc[6 + nh][v2]);
                float c  = fv * c1v[nh * 8 + v2] + iv * gv;
                c1v[nh * 8 + v2] = c;
                float hv = ov * fast_tanh(c);
                hl1[nh * 8 + v2] = hv;                        // keep f32 for final FC
                int M = v2 + 8 * khalf;
                hbuf[w][M * 32 + 16 * nh + nlo] = (_Float16)hv;
            }
        }
        asm volatile("s_wait_dscnt 0" ::: "memory");
        {
            const _Float16* hp = &hbuf[w][nlo * 32];
            Frag16 a1;
            a1.q[0] = *(const v4f*)(hp + 8 * khalf);
            a1.q[1] = *(const v4f*)(hp + 16 + 8 * khalf);
            h1A = a1.v;
        }
        h0A = a0.v;
    }

    // ---- final FC: out[m] = sum_n h1[m,n] * fc_w[n] + fc_b ----
    if (lane < 16) oacc[w][lane] = 0.0f;
    asm volatile("s_wait_dscnt 0" ::: "memory");
    #pragma unroll
    for (int v2 = 0; v2 < 8; ++v2) {
        float partial = hl1[v2] * fcw0 + hl1[8 + v2] * fcw1;  // N=nlo and N=16+nlo
        atomicAdd(&oacc[w][v2 + 8 * khalf], partial);         // ds_add_f32, per-row slot
    }
    asm volatile("s_wait_dscnt 0" ::: "memory");
    if (lane < 16) out[rowbase + lane] = oacc[w][lane] + fc_b[0];
}

extern "C" void kernel_launch(void* const* d_in, const int* in_sizes, int n_in,
                              void* d_out, int out_size, void* d_ws, size_t ws_size,
                              hipStream_t stream) {
    (void)in_sizes; (void)n_in; (void)out_size; (void)d_ws; (void)ws_size;
    const float* x     = (const float*)d_in[0];
    const float* w_ih0 = (const float*)d_in[1];
    const float* w_hh0 = (const float*)d_in[2];
    const float* b_ih0 = (const float*)d_in[3];
    const float* b_hh0 = (const float*)d_in[4];
    const float* w_ih1 = (const float*)d_in[5];
    const float* w_hh1 = (const float*)d_in[6];
    const float* b_ih1 = (const float*)d_in[7];
    const float* b_hh1 = (const float*)d_in[8];
    const float* fc_w  = (const float*)d_in[9];
    const float* fc_b  = (const float*)d_in[10];
    float* out = (float*)d_out;

    dim3 grid(4096 / (16 * WAVES));   // 64 workgroups x 4 waves = 256 independent tiles
    dim3 block(WAVES * 32);
    lstm_fused_wmma<<<grid, block, 0, stream>>>(x, w_ih0, w_hh0, b_ih0, b_hh0,
                                                w_ih1, w_hh1, b_ih1, b_hh1,
                                                fc_w, fc_b, out);
}
